// nosepExpert_68436008894761
// MI455X (gfx1250) — compile-verified, matrix operand order checked by reading, not run on hardware
//
#include <hip/hip_runtime.h>
#include <cstdint>

// ---------------------------------------------------------------------------
// Batched-expert MLP (E=8): out = L2norm_row( mean_e( relu(x@W1_e+b1_e)@W2_e + b2_e ) )
// N=262144, IN=256, HID=192, OUT=128.  Compute-bound => bf16 WMMA.
// v3: 32 rows per wave (two A tiles). Each LDS B-fragment read feeds TWO
// WMMAs -> halves LDS bandwidth demand (matches 256 B/clk WGP LDS roofline)
// and gives two independent accumulator chains to hide WMMA->WMMA hazards.
// Expert weights staged per-block via Tensor Data Mover (tensor_load_to_lds).
// ---------------------------------------------------------------------------

#define NROWS   262144
#define IN_DIM  256
#define HID_DIM 192
#define OUT_DIM 128
#define NEXP    8

#define KC1 8   // IN/32  K-chunks for GEMM1
#define HT1 12  // HID/16 N-tiles  for GEMM1
#define KC2 6   // HID/32 K-chunks for GEMM2
#define NT2 8   // OUT/16 N-tiles  for GEMM2

// per-expert contiguous fragment pack in d_ws (bf16 elements):
//   [ W1 frags: KC1*HT1*512 = 49152 ][ W2 frags: KC2*NT2*512 = 24576 ] = 73728
#define EXP_ELEMS   73728          // bf16 per expert  (147456 B)
#define W2_OFF      49152          // bf16 offset of W2 frags within pack
#define EXP_BYTES   147456
#define EXP_QWORDS  18432          // 8-byte elements per pack (for TDM)

#define LDS_STRIDE  200            // h row stride in bf16 (400 B, 16B aligned)
#define ROWS_PER_WAVE 32
#define H_STRIP     (ROWS_PER_WAVE * LDS_STRIDE)     // bf16 per wave strip
#define LDS_W_BYTES EXP_BYTES                        // 147456
#define LDS_H_BYTES (8 * H_STRIP * 2)                // 102400
#define LDS_TOTAL   (LDS_W_BYTES + LDS_H_BYTES)      // 249856 < 320KB WGP LDS

typedef __bf16 bf16;
typedef bf16     v16bf __attribute__((ext_vector_type(16)));
typedef float    v8f   __attribute__((ext_vector_type(8)));
typedef unsigned u32x4 __attribute__((ext_vector_type(4)));
typedef int      i32x8 __attribute__((ext_vector_type(8)));
typedef int      i32x4 __attribute__((ext_vector_type(4)));

__device__ __forceinline__ bf16 f2bf(float f) {
    unsigned u = __float_as_uint(f);
    u += 0x7FFFu + ((u >> 16) & 1u);          // round-to-nearest-even
    return __builtin_bit_cast(bf16, (unsigned short)(u >> 16));
}

// ---------------------------------------------------------------------------
// Prep: fp32 weights -> bf16 B-fragments (ISA 7.12.2 layout), packed
// per-expert contiguous so one TDM descriptor covers W1+W2 of an expert.
// ---------------------------------------------------------------------------
__global__ void __launch_bounds__(256)
prep_weights(const float* __restrict__ W1, const float* __restrict__ W2,
             bf16* __restrict__ fW) {
    const int lane = threadIdx.x & 31;
    const int tile = (blockIdx.x * blockDim.x + threadIdx.x) >> 5;
    const int n    = lane & 15;
    const int half = lane >> 4;
    const int T1   = NEXP * KC1 * HT1;      // 768
    const int T2   = NEXP * KC2 * NT2;      // 384

    if (tile < T1) {
        int e  = tile / (KC1 * HT1);
        int kc = (tile / HT1) % KC1;
        int ht = tile % HT1;
        const float* src = W1 + (size_t)e * IN_DIM * HID_DIM + ht * 16 + n;
        v16bf v;
#pragma unroll
        for (int j = 0; j < 16; ++j) {
            int k = kc * 32 + half * 8 + (j < 8 ? j : j + 8);
            v[j] = f2bf(src[(size_t)k * HID_DIM]);
        }
        *(v16bf*)(fW + (size_t)e * EXP_ELEMS + (kc * HT1 + ht) * 512 + lane * 16) = v;
    } else if (tile < T1 + T2) {
        int t  = tile - T1;
        int e  = t / (KC2 * NT2);
        int kc = (t / NT2) % KC2;
        int nt = t % NT2;
        const float* src = W2 + (size_t)e * HID_DIM * OUT_DIM + nt * 16 + n;
        v16bf v;
#pragma unroll
        for (int j = 0; j < 16; ++j) {
            int k = kc * 32 + half * 8 + (j < 8 ? j : j + 8);
            v[j] = f2bf(src[(size_t)k * OUT_DIM]);
        }
        *(v16bf*)(fW + (size_t)e * EXP_ELEMS + W2_OFF + (kc * NT2 + nt) * 512 + lane * 16) = v;
    }
}

// ---------------------------------------------------------------------------
// Main kernel: 256 threads = 8 waves; each wave owns 32 rows (2 M-tiles).
// Per expert: TDM stages the 144KB fragment pack into LDS; every B fragment
// register-load then feeds 2 WMMAs (one per M-tile).
// ---------------------------------------------------------------------------
__global__ void __launch_bounds__(256)
moe_mlp_kernel(const float* __restrict__ x,
               const float* __restrict__ b1,
               const float* __restrict__ b2,
               const bf16*  __restrict__ fW,
               float* __restrict__ out) {
    extern __shared__ __align__(16) char smem[];
    bf16* lw = (bf16*)smem;                          // weights (144 KB)
    bf16* lh = (bf16*)(smem + LDS_W_BYTES);          // h strips (100 KB)

    const int lane = threadIdx.x & 31;
    const int wave = threadIdx.x >> 5;
    const int n16  = lane & 15;
    const int half = lane >> 4;
    const int row0 = (blockIdx.x * 8 + wave) * ROWS_PER_WAVE;
    bf16* h = lh + wave * H_STRIP;                   // private per-wave strip

    // ---- x A-fragments for both M-tiles (built once, reused 8x) ----
    v16bf xa[2][KC1];
#pragma unroll
    for (int t = 0; t < 2; ++t) {
        const float* xr = x + (size_t)(row0 + t * 16 + n16) * IN_DIM;
#pragma unroll
        for (int kc = 0; kc < KC1; ++kc) {
            int kb = kc * 32 + half * 8;
            float4 f0 = *(const float4*)(xr + kb);
            float4 f1 = *(const float4*)(xr + kb + 4);
            float4 f2 = *(const float4*)(xr + kb + 16);
            float4 f3 = *(const float4*)(xr + kb + 20);
            v16bf a;
            a[0]  = f2bf(f0.x); a[1]  = f2bf(f0.y); a[2]  = f2bf(f0.z); a[3]  = f2bf(f0.w);
            a[4]  = f2bf(f1.x); a[5]  = f2bf(f1.y); a[6]  = f2bf(f1.z); a[7]  = f2bf(f1.w);
            a[8]  = f2bf(f2.x); a[9]  = f2bf(f2.y); a[10] = f2bf(f2.z); a[11] = f2bf(f2.w);
            a[12] = f2bf(f3.x); a[13] = f2bf(f3.y); a[14] = f2bf(f3.z); a[15] = f2bf(f3.w);
            xa[t][kc] = a;
        }
    }

    v8f acc[2][NT2];
#pragma unroll
    for (int t = 0; t < 2; ++t)
#pragma unroll
        for (int nt = 0; nt < NT2; ++nt) acc[t][nt] = (v8f){};

#pragma unroll 1
    for (int e = 0; e < NEXP; ++e) {
        __syncthreads();   // previous expert's LDS weights no longer in use

#if __has_builtin(__builtin_amdgcn_tensor_load_to_lds)
        if (wave == 0) {
            // ---- Tensor Data Mover: 1-D copy of the 144KB pack to LDS ----
            unsigned long long ga =
                (unsigned long long)(uintptr_t)fW + (unsigned long long)e * EXP_BYTES;
            unsigned lds_off = (unsigned)(uintptr_t)lw;   // flat LDS addr[31:0] = offset
            u32x4 g0;
            g0[0] = 1u;                                   // count=1 (valid user D#)
            g0[1] = lds_off;                              // lds_addr
            g0[2] = (unsigned)ga;                         // global_addr[31:0]
            g0[3] = (unsigned)((ga >> 32) & 0x01FFFFFFu)  // global_addr[56:32]
                    | 0x80000000u;                        // type=2 ("image")
            i32x8 g1;
            g1[0] = 0x00030000;                 // wg_mask=0 | data_size=3 (8B)
            g1[1] = (int)(EXP_QWORDS << 16);    // tensor_dim0[15:0] << 16
            g1[2] = 0x00010000;                 // tensor_dim0[31:16]=0 | tensor_dim1=1
            g1[3] = (int)(EXP_QWORDS << 16);    // tensor_dim1 hi=0 | tile_dim0=18432
            g1[4] = 0;                          // tile_dim1=0 (unused), tile_dim2=0
            g1[5] = EXP_QWORDS;                 // tensor_dim0_stride[31:0]
            g1[6] = 0;
            g1[7] = 0;
            i32x4 z4 = {0, 0, 0, 0};
            i32x8 z8 = {0, 0, 0, 0, 0, 0, 0, 0};
            __builtin_amdgcn_tensor_load_to_lds(g0, g1, z4, z4, z8, 0);
            __builtin_amdgcn_s_wait_tensorcnt(0);
        }
#else
        {   // fallback: cooperative VGPR copy global -> LDS
            const uint4* gsrc = (const uint4*)(fW + (size_t)e * EXP_ELEMS);
            uint4* ldst = (uint4*)lw;
            for (int i = threadIdx.x; i < EXP_BYTES / 16; i += 256)
                ldst[i] = gsrc[i];
        }
#endif
        __syncthreads();   // weights visible to all waves

        // ===== GEMM1: h = relu(x @ W1_e + b1_e)  (2 M-tiles per B frag) =====
#pragma unroll 1
        for (int ht = 0; ht < HT1; ++ht) {
            v8f c0 = (v8f){};
            v8f c1 = (v8f){};
#pragma unroll
            for (int kc = 0; kc < KC1; ++kc) {
                v16bf b = *(const v16bf*)(lw + (kc * HT1 + ht) * 512 + lane * 16);
                c0 = __builtin_amdgcn_wmma_f32_16x16x32_bf16(
                         false, xa[0][kc], false, b, (short)0, c0, false, false);
                c1 = __builtin_amdgcn_wmma_f32_16x16x32_bf16(
                         false, xa[1][kc], false, b, (short)0, c1, false, false);
            }
            float bias = b1[e * HID_DIM + ht * 16 + n16];
#pragma unroll
            for (int r = 0; r < 8; ++r) {
                float v0 = c0[r] + bias;          // C-layout: row = half*8+r, col = n16
                float v1 = c1[r] + bias;
                v0 = v0 > 0.0f ? v0 : 0.0f;
                v1 = v1 > 0.0f ? v1 : 0.0f;
                h[(half * 8 + r) * LDS_STRIDE + ht * 16 + n16]        = f2bf(v0);
                h[(16 + half * 8 + r) * LDS_STRIDE + ht * 16 + n16]   = f2bf(v1);
            }
        }

        // ===== GEMM2: acc += h @ W2_e  (2 M-tiles per B frag) =====
#pragma unroll 1
        for (int kc = 0; kc < KC2; ++kc) {
            int kb = kc * 32 + half * 8;
            union { v16bf v; uint4 q[2]; } ha0, ha1;
            ha0.q[0] = *(const uint4*)(h + n16 * LDS_STRIDE + kb);
            ha0.q[1] = *(const uint4*)(h + n16 * LDS_STRIDE + kb + 16);
            ha1.q[0] = *(const uint4*)(h + (16 + n16) * LDS_STRIDE + kb);
            ha1.q[1] = *(const uint4*)(h + (16 + n16) * LDS_STRIDE + kb + 16);
#pragma unroll
            for (int nt = 0; nt < NT2; ++nt) {
                v16bf b = *(const v16bf*)(lw + W2_OFF + (kc * NT2 + nt) * 512 + lane * 16);
                acc[0][nt] = __builtin_amdgcn_wmma_f32_16x16x32_bf16(
                                 false, ha0.v, false, b, (short)0, acc[0][nt], false, false);
                acc[1][nt] = __builtin_amdgcn_wmma_f32_16x16x32_bf16(
                                 false, ha1.v, false, b, (short)0, acc[1][nt], false, false);
            }
        }
    }

    // ---- epilogue: mean (incl. mean of b2), L2 normalize rows, store ----
    float bsum[NT2];
#pragma unroll
    for (int nt = 0; nt < NT2; ++nt) {
        float bs = 0.0f;
#pragma unroll
        for (int e = 0; e < NEXP; ++e) bs += b2[e * OUT_DIM + nt * 16 + n16];
        bsum[nt] = bs;
    }

#pragma unroll
    for (int t = 0; t < 2; ++t) {
#pragma unroll
        for (int nt = 0; nt < NT2; ++nt)
#pragma unroll
            for (int r = 0; r < 8; ++r)
                acc[t][nt][r] = (acc[t][nt][r] + bsum[nt]) * 0.125f;

#pragma unroll
        for (int r = 0; r < 8; ++r) {
            float s = 0.0f;
#pragma unroll
            for (int nt = 0; nt < NT2; ++nt) s += acc[t][nt][r] * acc[t][nt][r];
            // butterfly within each 16-lane half (row lives on one half, wave32)
            s += __shfl_xor(s, 1);
            s += __shfl_xor(s, 2);
            s += __shfl_xor(s, 4);
            s += __shfl_xor(s, 8);
            float inv = rsqrtf(s);
            float* orow = out + (size_t)(row0 + t * 16 + half * 8 + r) * OUT_DIM;
#pragma unroll
            for (int nt = 0; nt < NT2; ++nt)
                orow[nt * 16 + n16] = acc[t][nt][r] * inv;
        }
    }
}

// ---------------------------------------------------------------------------
extern "C" void kernel_launch(void* const* d_in, const int* in_sizes, int n_in,
                              void* d_out, int out_size, void* d_ws, size_t ws_size,
                              hipStream_t stream) {
    const float* x  = (const float*)d_in[0];
    const float* W1 = (const float*)d_in[1];
    const float* b1 = (const float*)d_in[2];
    const float* W2 = (const float*)d_in[3];
    const float* b2 = (const float*)d_in[4];
    float* out = (float*)d_out;

    bf16* fW = (bf16*)d_ws;   // 8 experts * 147456 B = 1.18 MB fragment packs

    // 768 + 384 = 1152 weight tiles, one wave each -> 144 blocks of 256
    prep_weights<<<144, 256, 0, stream>>>(W1, W2, fW);

    // 32 rows per wave, 8 waves per block -> 256 rows/block, 1024 blocks
    moe_mlp_kernel<<<NROWS / 256, 256, LDS_TOTAL, stream>>>(x, b1, b2, fW, out);
}
